// AdaptCores_and_Test_Individual_28003186770538
// MI455X (gfx1250) — matrix operand order repaired and checked by reading, not compiled, not run on hardware
//
#include <hip/hip_runtime.h>
#include <hip/hip_bf16.h>

typedef __attribute__((ext_vector_type(16))) __bf16 v16bf;
typedef __attribute__((ext_vector_type(8)))  __bf16 v8bf;
typedef __attribute__((ext_vector_type(8)))  float  v8f;
typedef __attribute__((ext_vector_type(4)))  float  f32x4;

#define DIM   1024
#define MROWS 16384   // B*S = 8*2048
#define KC    32      // K step
#define MT    128     // block tile M
#define NT    64      // block tile N
#define LDX   36      // padded LDS row stride for X tile (fp32 elements)
#define LDB   40      // padded LDS row stride for B tiles (bf16 elements)
#define NIT   (DIM / KC)

static __device__ __forceinline__ __bf16 f2bf(float f) { return (__bf16)f; }

// LDS byte offset of a shared-memory pointer (addrspace(3) offsets start at 0;
// the generic address's low 32 bits are the LDS offset).
static __device__ __forceinline__ uint32_t lds_off(const void* p) {
  return (uint32_t)(uintptr_t)p;
}

// gfx1250 async DMA: LDS[vdst_lane] = MEM[vaddr_lane], 16B per lane, ASYNCcnt.
static __device__ __forceinline__ void async_copy_b128(uint32_t lds, const void* g) {
  asm volatile("global_load_async_to_lds_b128 %0, %1, off"
               :: "v"(lds), "v"((uint64_t)(uintptr_t)g) : "memory");
}
static __device__ __forceinline__ void wait_async0() {
  asm volatile("s_wait_asynccnt 0" ::: "memory");
}

// ---------------------------------------------------------------------------
// Kernel 1: collapse TT cores into M (1024x8) and N (8x1024).
//   M[d,p] = sum_{r1,r2} core0[0,i1,r1]*core1[r1,i2,r2]*core2[r2,i3,p]
//   N[q,o] = sum_{r,s}   core3[q,n1,r]*core4[r,n2,s]*core5[s,n3,0]
// d = i3*128 + i2*8 + i1 ; o = n1*128 + n2*8 + n3
// ---------------------------------------------------------------------------
__global__ void prep_mn_kernel(const float* __restrict__ c0, const float* __restrict__ c1,
                               const float* __restrict__ c2, const float* __restrict__ c3,
                               const float* __restrict__ c4, const float* __restrict__ c5,
                               float* __restrict__ Mw, float* __restrict__ Nw) {
  const int t = threadIdx.x;  // 256 threads
  for (int d = t; d < DIM; d += 256) {
    const int i1 = d & 7, i2 = (d >> 3) & 15, i3 = d >> 7;
    float tr[8];
#pragma unroll
    for (int r2 = 0; r2 < 8; ++r2) {
      float s = 0.f;
#pragma unroll
      for (int r1 = 0; r1 < 8; ++r1)
        s += c0[i1 * 8 + r1] * c1[(r1 * 16 + i2) * 8 + r2];
      tr[r2] = s;
    }
#pragma unroll
    for (int p = 0; p < 8; ++p) {
      float s = 0.f;
#pragma unroll
      for (int r2 = 0; r2 < 8; ++r2)
        s += tr[r2] * c2[(r2 * 8 + i3) * 8 + p];
      Mw[d * 8 + p] = s;
    }
  }
  for (int o = t; o < DIM; o += 256) {
    const int n3 = o & 7, n2 = (o >> 3) & 15, n1 = o >> 7;
    float v[8];
#pragma unroll
    for (int r = 0; r < 8; ++r) {
      float s = 0.f;
#pragma unroll
      for (int ss = 0; ss < 8; ++ss)
        s += c4[(r * 16 + n2) * 8 + ss] * c5[ss * 8 + n3];
      v[r] = s;
    }
#pragma unroll
    for (int q = 0; q < 8; ++q) {
      float s = 0.f;
#pragma unroll
      for (int r = 0; r < 8; ++r)
        s += c3[(q * 8 + n1) * 8 + r] * v[r];
      Nw[q * DIM + o] = s;
    }
  }
}

// ---------------------------------------------------------------------------
// Kernel 2: W_eff[o,d] = W[o,d] + 16 * sum_q M[d,q]*N[q,o]; split to bf16 hi/lo.
// ---------------------------------------------------------------------------
__global__ void prep_weff_kernel(const float* __restrict__ W, const float* __restrict__ Mw,
                                 const float* __restrict__ Nw,
                                 __bf16* __restrict__ Whi, __bf16* __restrict__ Wlo) {
  const int idx = blockIdx.x * 256 + threadIdx.x;  // idx = o*1024 + d
  const int o = idx >> 10, d = idx & (DIM - 1);
  float acc = 0.f;
#pragma unroll
  for (int q = 0; q < 8; ++q) acc += Mw[d * 8 + q] * Nw[q * DIM + o];
  const float w = W[idx] + 16.0f * acc;
  const __bf16 hi = f2bf(w);
  Whi[idx] = hi;
  Wlo[idx] = f2bf(w - (float)hi);
}

// ---------------------------------------------------------------------------
// Kernel 3: out[m,o] = sum_d x[m,d]*W_eff[o,d] + b[o].
// bf16x3 split-precision WMMA, double-buffered GLOBAL_LOAD_ASYNC_TO_LDS_B128.
// Block = 8 waves, tile 128(M) x 64(N); wave w: rows [w*16,w*16+16) x 64 cols.
// ---------------------------------------------------------------------------
__global__ __launch_bounds__(256) void gemm_kernel(const float* __restrict__ X,
                                                   const __bf16* __restrict__ Whi,
                                                   const __bf16* __restrict__ Wlo,
                                                   const float* __restrict__ bias,
                                                   float* __restrict__ out) {
  // Manual LDS layout (all 16B aligned):
  //   sX  : 2 x [128][36] fp32  = 36864 B
  //   sBhi: 2 x [64][40]  bf16  = 10240 B
  //   sBlo: 2 x [64][40]  bf16  = 10240 B
  __shared__ __align__(16) char smem[36864 + 10240 + 10240];
  float*  const sX   = (float*)smem;
  __bf16* const sBhi = (__bf16*)(smem + 36864);
  __bf16* const sBlo = (__bf16*)(smem + 36864 + 10240);

  const int tid  = threadIdx.x;
  const int wave = tid >> 5;
  const int lane = tid & 31;
  const int half = lane >> 4;   // K-half selector per ISA A/B fragment layouts
  const int l16  = lane & 15;
  const int rowBase = blockIdx.x * MT;
  const int colBase = blockIdx.y * NT;

  // cooperative async-copy indices
  const int ar = tid >> 3;            // 0..31: X row group (4 rows per thread)
  const int ac = (tid & 7) * 4;       // X float4 column within K-step
  const int bn = tid >> 2;            // 0..63: W_eff row (output col)
  const int bc = (tid & 3) * 8;       // W v8bf column within K-step

  // Issue the 6 async B128 copies for K-tile `kb` into buffer `buf`.
  auto issue_tile = [&](int kb, int buf) {
    float* dstX = sX + buf * (MT * LDX);
#pragma unroll
    for (int r = 0; r < MT; r += 32) {
      const int row = r + ar;
      async_copy_b128(lds_off(dstX + row * LDX + ac),
                      X + (size_t)(rowBase + row) * DIM + kb + ac);
    }
    const size_t g = (size_t)(colBase + bn) * DIM + kb + bc;
    async_copy_b128(lds_off(sBhi + buf * (NT * LDB) + bn * LDB + bc), Whi + g);
    async_copy_b128(lds_off(sBlo + buf * (NT * LDB) + bn * LDB + bc), Wlo + g);
  };

  v8f acc[4];
#pragma unroll
  for (int i = 0; i < 4; ++i) acc[i] = (v8f)(0.0f);

  issue_tile(0, 0);  // prologue

  for (int it = 0; it < NIT; ++it) {
    const int cur = it & 1;
    // Own async copies for `cur` complete -> barrier makes them visible to all.
    wait_async0();
    __syncthreads();
    if (it + 1 < NIT) issue_tile((it + 1) * KC, cur ^ 1);  // overlap with compute

    // Build A fragments (16x32): fp32 from LDS, split to bf16 hi/lo in-register.
    // Element e<8 -> K = half*8+e ; e>=8 -> K = 16 + half*8 + (e-8).
    const float* xr = sX + cur * (MT * LDX) + (wave * 16 + l16) * LDX;
    f32x4 xc[4];
    xc[0] = *(const f32x4*)(xr + half * 8);
    xc[1] = *(const f32x4*)(xr + half * 8 + 4);
    xc[2] = *(const f32x4*)(xr + 16 + half * 8);
    xc[3] = *(const f32x4*)(xr + 16 + half * 8 + 4);
    v16bf ahi, alo;
#pragma unroll
    for (int c = 0; c < 4; ++c) {
#pragma unroll
      for (int j = 0; j < 4; ++j) {
        const float v = xc[c][j];
        const __bf16 h = f2bf(v);
        ahi[c * 4 + j] = h;
        alo[c * 4 + j] = f2bf(v - (float)h);
      }
    }

#pragma unroll
    for (int nt = 0; nt < 4; ++nt) {
      // B fragment (32x16): N = l16, e<8 -> K = half*8+e ; e>=8 -> K = 16 + half*8 + (e-8).
      const __bf16* bh = sBhi + cur * (NT * LDB) + (nt * 16 + l16) * LDB + half * 8;
      const __bf16* bl = sBlo + cur * (NT * LDB) + (nt * 16 + l16) * LDB + half * 8;
      v8bf b0 = *(const v8bf*)(bh);
      v8bf b1 = *(const v8bf*)(bh + 16);
      v8bf m0 = *(const v8bf*)(bl);
      v8bf m1 = *(const v8bf*)(bl + 16);
      v16bf bhi, blo;
#pragma unroll
      for (int i = 0; i < 8; ++i) {
        bhi[i] = b0[i]; bhi[8 + i] = b1[i];
        blo[i] = m0[i]; blo[8 + i] = m1[i];
      }
      // bf16x3: hi*hi + hi*lo + lo*hi (lo*lo term ~2^-18, dropped)
      acc[nt] = __builtin_amdgcn_wmma_f32_16x16x32_bf16(
          false, ahi, false, bhi, (short)0, acc[nt], false, false);
      acc[nt] = __builtin_amdgcn_wmma_f32_16x16x32_bf16(
          false, ahi, false, blo, (short)0, acc[nt], false, false);
      acc[nt] = __builtin_amdgcn_wmma_f32_16x16x32_bf16(
          false, alo, false, bhi, (short)0, acc[nt], false, false);
    }
    __syncthreads();  // all reads of `cur` done before it is refilled (it+1 issues cur^1, it+2 issues cur)
  }

  // Epilogue: C/D layout — VGPR j: lanes 0-15 row j, lanes 16-31 row j+8; col = l16.
  const int r0 = rowBase + wave * 16 + half * 8;
#pragma unroll
  for (int nt = 0; nt < 4; ++nt) {
    const int col = colBase + nt * 16 + l16;
    const float bv = bias[col];
#pragma unroll
    for (int j = 0; j < 8; ++j) {
      out[(size_t)(r0 + j) * DIM + col] = acc[nt][j] + bv;
    }
  }
}

// ---------------------------------------------------------------------------
extern "C" void kernel_launch(void* const* d_in, const int* in_sizes, int n_in,
                              void* d_out, int out_size, void* d_ws, size_t ws_size,
                              hipStream_t stream) {
  (void)in_sizes; (void)n_in; (void)out_size; (void)ws_size;
  const float* x  = (const float*)d_in[0];
  const float* W  = (const float*)d_in[1];
  const float* b  = (const float*)d_in[2];
  const float* c0 = (const float*)d_in[3];
  const float* c1 = (const float*)d_in[4];
  const float* c2 = (const float*)d_in[5];
  const float* c3 = (const float*)d_in[6];
  const float* c4 = (const float*)d_in[7];
  const float* c5 = (const float*)d_in[8];
  float* out = (float*)d_out;

  // workspace layout: M (32KB) | N (32KB) | Whi (2MB) | Wlo (2MB)
  char* ws = (char*)d_ws;
  float*  Mw  = (float*)ws;
  float*  Nw  = (float*)(ws + 32 * 1024);
  __bf16* Whi = (__bf16*)(ws + 64 * 1024);
  __bf16* Wlo = (__bf16*)(ws + 64 * 1024 + 2 * 1024 * 1024);

  prep_mn_kernel<<<1, 256, 0, stream>>>(c0, c1, c2, c3, c4, c5, Mw, Nw);
  prep_weff_kernel<<<(DIM * DIM) / 256, 256, 0, stream>>>(W, Mw, Nw, Whi, Wlo);
  dim3 grid(MROWS / MT, DIM / NT);
  gemm_kernel<<<grid, 256, 0, stream>>>(x, Whi, Wlo, b, out);
}
